// WhisperDecoderCoreKVCache_40827959116095
// MI455X (gfx1250) — matrix operand order repaired and checked by reading, not compiled, not run on hardware
//
#include <hip/hip_runtime.h>
#include <hip/hip_bf16.h>

#define L_  12
#define B2_ 2
#define D_  768
#define H_  12
#define HD_ 64
#define ENC_ 1500
#define MC_  448
#define V_   51865

typedef __bf16 bf16_t;
typedef bf16_t v16bf __attribute__((ext_vector_type(16)));
typedef bf16_t v8bf  __attribute__((ext_vector_type(8)));
typedef float  v8f   __attribute__((ext_vector_type(8)));

__device__ __forceinline__ bf16_t f2bf(float f) {
  union { float f; unsigned u; } c; c.f = f;
  unsigned r = c.u + 0x7FFFu + ((c.u >> 16) & 1u);   // round-to-nearest-even
  unsigned short h = (unsigned short)(r >> 16);
  return __builtin_bit_cast(bf16_t, h);
}

// ---- f32 -> bf16 pre-conversion (vectorized, n divisible by 4) ----
__global__ __launch_bounds__(256)
void cvt_bf16_kernel(const float* __restrict__ src, bf16_t* __restrict__ dst, int n4) {
  int i = blockIdx.x * 256 + threadIdx.x;
  if (i >= n4) return;
  float4 v = ((const float4*)src)[i];
  union { bf16_t b[4]; uint2 u; } o;
  o.b[0] = f2bf(v.x); o.b[1] = f2bf(v.y); o.b[2] = f2bf(v.z); o.b[3] = f2bf(v.w);
  ((uint2*)dst)[i] = o.u;
}

// -------------------------------------------------------------------------
// FAST PATH: C[M,N] = X[M,K] @ W[N,K]^T (+bias), X/W already bf16.
// grid = (N/64, ceil(M/128)), block = 256 (8 waves). Wave: 16(M) x 64(N).
// Inner loop: pure b128 loads + v_wmma (no conversion VALU).
// -------------------------------------------------------------------------
__global__ __launch_bounds__(256)
void wmma_gemm_bf16(const bf16_t* __restrict__ X, const bf16_t* __restrict__ W,
                    const float* __restrict__ bias, float* __restrict__ C,
                    int M, int N, int K) {
  const int lane = threadIdx.x & 31;
  const int wave = threadIdx.x >> 5;
  const int m0 = blockIdx.y * 128 + wave * 16;
  const int n0 = blockIdx.x * 64;
  const int halfSel = lane >> 4;
  const int lrow = lane & 15;
  const int kbA = halfSel * 16;      // A: lanes 0-15 hold K 0-15, lanes 16-31 hold K 16-31
  const int kbB = halfSel * 16;      // B: lanes 0-15 hold K 0-15, lanes 16-31 hold K 16-31

  int row = m0 + lrow;
  int rowL = row < M ? row : (M - 1);
  const bf16_t* xrow = X + (size_t)rowL * K;

  const bf16_t* wr[4];
  float bv[4];
#pragma unroll
  for (int t = 0; t < 4; ++t) {
    int nt = n0 + t * 16 + lrow;
    wr[t] = W + (size_t)nt * K;
    bv[t] = bias ? bias[nt] : 0.f;
  }

  v8f acc[4] = {};
  for (int kk = 0; kk < K; kk += 32) {
    v8bf lo = *(const v8bf*)(xrow + kk + kbA);
    v8bf hi = *(const v8bf*)(xrow + kk + kbA + 8);
    v16bf a = __builtin_shufflevector(lo, hi,
        0, 1, 2, 3, 4, 5, 6, 7, 8, 9, 10, 11, 12, 13, 14, 15);
#pragma unroll
    for (int t = 0; t < 4; ++t) {
      v16bf bfr = *(const v16bf*)(wr[t] + kk + kbB);
      acc[t] = __builtin_amdgcn_wmma_f32_16x16x32_bf16(
          false, a, false, bfr, (short)0, acc[t], false, false);
    }
  }

  const int rowBase = m0 + halfSel * 8;
#pragma unroll
  for (int t = 0; t < 4; ++t) {
    int col = n0 + t * 16 + lrow;
#pragma unroll
    for (int r = 0; r < 8; ++r) {
      int rw = rowBase + r;
      if (rw < M) C[(size_t)rw * N + col] = acc[t][r] + bv[t];
    }
  }
}

// -------------------------------------------------------------------------
// FALLBACK (small ws): same GEMM with fused f32->bf16 conversion.
// -------------------------------------------------------------------------
__global__ __launch_bounds__(256)
void wmma_gemm_f32src(const float* __restrict__ X, const float* __restrict__ W,
                      const float* __restrict__ bias, float* __restrict__ C,
                      int M, int N, int K) {
  const int lane = threadIdx.x & 31;
  const int wave = threadIdx.x >> 5;
  const int m0 = blockIdx.y * 128 + wave * 16;
  const int n0 = blockIdx.x * 64;
  const int halfSel = lane >> 4;
  const int lrow = lane & 15;
  const int kbA = halfSel * 16;
  const int kbB = halfSel * 16;

  int row = m0 + lrow;
  int rowL = row < M ? row : (M - 1);
  const float* xrow = X + (size_t)rowL * K;

  const float* wr[4];
  float bv[4];
#pragma unroll
  for (int t = 0; t < 4; ++t) {
    int nt = n0 + t * 16 + lrow;
    wr[t] = W + (size_t)nt * K;
    bv[t] = bias ? bias[nt] : 0.f;
  }

  v8f acc[4] = {};
  for (int kk = 0; kk < K; kk += 32) {
    v16bf a;
    {
      const float4* p = (const float4*)(xrow + kk + kbA);
      float4 x0 = p[0], x1 = p[1], x2 = p[2], x3 = p[3];
      a[0]=f2bf(x0.x);  a[1]=f2bf(x0.y);  a[2]=f2bf(x0.z);  a[3]=f2bf(x0.w);
      a[4]=f2bf(x1.x);  a[5]=f2bf(x1.y);  a[6]=f2bf(x1.z);  a[7]=f2bf(x1.w);
      a[8]=f2bf(x2.x);  a[9]=f2bf(x2.y);  a[10]=f2bf(x2.z); a[11]=f2bf(x2.w);
      a[12]=f2bf(x3.x); a[13]=f2bf(x3.y); a[14]=f2bf(x3.z); a[15]=f2bf(x3.w);
    }
#pragma unroll
    for (int t = 0; t < 4; ++t) {
      const float4* p = (const float4*)(wr[t] + kk + kbB);
      float4 w0 = p[0], w1 = p[1], w2 = p[2], w3 = p[3];
      v16bf bfr;
      bfr[0]=f2bf(w0.x);  bfr[1]=f2bf(w0.y);  bfr[2]=f2bf(w0.z);  bfr[3]=f2bf(w0.w);
      bfr[4]=f2bf(w1.x);  bfr[5]=f2bf(w1.y);  bfr[6]=f2bf(w1.z);  bfr[7]=f2bf(w1.w);
      bfr[8]=f2bf(w2.x);  bfr[9]=f2bf(w2.y);  bfr[10]=f2bf(w2.z); bfr[11]=f2bf(w2.w);
      bfr[12]=f2bf(w3.x); bfr[13]=f2bf(w3.y); bfr[14]=f2bf(w3.z); bfr[15]=f2bf(w3.w);
      acc[t] = __builtin_amdgcn_wmma_f32_16x16x32_bf16(
          false, a, false, bfr, (short)0, acc[t], false, false);
    }
  }

  const int rowBase = m0 + halfSel * 8;
#pragma unroll
  for (int t = 0; t < 4; ++t) {
    int col = n0 + t * 16 + lrow;
#pragma unroll
    for (int r = 0; r < 8; ++r) {
      int rw = rowBase + r;
      if (rw < M) C[(size_t)rw * N + col] = acc[t][r] + bv[t];
    }
  }
}

// ---- GEMV: y[b,n] = dot(x[b,:], W[n,:]) + bias (+gelu) (+resid). ----
__global__ __launch_bounds__(256)
void gemv_kernel(const float* __restrict__ x, const float* __restrict__ W,
                 const float* __restrict__ bias, const float* __restrict__ resid,
                 float* __restrict__ y, int N, int K, int gelu) {
  const int lane = threadIdx.x & 31;
  const int wave = threadIdx.x >> 5;
  const int n = blockIdx.x * 8 + wave;
  const int b = blockIdx.y;
  if (n >= N) return;
  const float* xp = x + (size_t)b * K;
  const float* wp = W + (size_t)n * K;
  float s = 0.f;
  for (int k = lane * 4; k < K; k += 128) {
    float4 xv = *(const float4*)(xp + k);
    float4 wv = *(const float4*)(wp + k);
    s += xv.x * wv.x + xv.y * wv.y + xv.z * wv.z + xv.w * wv.w;
  }
#pragma unroll
  for (int off = 16; off > 0; off >>= 1) s += __shfl_xor(s, off, 32);
  if (lane == 0) {
    float v = s + (bias ? bias[n] : 0.f);
    if (gelu) v = 0.5f * v * (1.f + erff(v * 0.70710678118f));
    if (resid) v += resid[(size_t)b * N + n];
    y[(size_t)b * N + n] = v;
  }
}

// ---- LayerNorm over last dim (D=768). grid = B, block = 256. ----
__global__ __launch_bounds__(256)
void ln_kernel(const float* __restrict__ x, const float* __restrict__ w,
               const float* __restrict__ bp, float* __restrict__ y) {
  __shared__ float red[256];
  const int b = blockIdx.x, tid = threadIdx.x;
  const float* xp = x + (size_t)b * D_;
  float s = 0.f;
  for (int i = tid; i < D_; i += 256) s += xp[i];
  red[tid] = s; __syncthreads();
  for (int o = 128; o > 0; o >>= 1) { if (tid < o) red[tid] += red[tid + o]; __syncthreads(); }
  float mean = red[0] * (1.f / D_);
  __syncthreads();
  float v = 0.f;
  for (int i = tid; i < D_; i += 256) { float d = xp[i] - mean; v += d * d; }
  red[tid] = v; __syncthreads();
  for (int o = 128; o > 0; o >>= 1) { if (tid < o) red[tid] += red[tid + o]; __syncthreads(); }
  float rstd = rsqrtf(red[0] * (1.f / D_) + 1e-5f);
  for (int i = tid; i < D_; i += 256)
    y[(size_t)b * D_ + i] = (xp[i] - mean) * rstd * w[i] + bp[i];
}

// ---- Single-query attention. grid = (H, B), block = 256. Scores in LDS. ----
__global__ __launch_bounds__(256)
void attn_kernel(const float* __restrict__ q, const float* __restrict__ kpast,
                 const float* __restrict__ vpast, const float* __restrict__ knew,
                 const float* __restrict__ vnew, const float* __restrict__ mask,
                 float* __restrict__ out, int S_past, int S_total) {
  __shared__ float sc[ENC_ + 4];
  __shared__ float red[256];
  __shared__ float qv[HD_];
  __shared__ float part[4][HD_];
  const int h = blockIdx.x, b = blockIdx.y, tid = threadIdx.x;
  const int d0 = h * HD_;
  if (tid < HD_) qv[tid] = q[(size_t)b * D_ + d0 + tid];
  __syncthreads();
  float lmax = -1e30f;
  for (int j = tid; j < S_total; j += 256) {
    const float* kp = (j < S_past) ? (kpast + ((size_t)b * S_past + j) * D_ + d0)
                                   : (knew + (size_t)b * D_ + d0);
    float dot = 0.f;
#pragma unroll
    for (int i = 0; i < HD_; i += 4) {
      float4 kv = *(const float4*)(kp + i);
      dot += qv[i] * kv.x + qv[i + 1] * kv.y + qv[i + 2] * kv.z + qv[i + 3] * kv.w;
    }
    float sv = dot * 0.125f + (mask ? mask[j] : 0.f);  // SCALE^2 = HD^-0.5 = 1/8
    sc[j] = sv;
    lmax = fmaxf(lmax, sv);
  }
  red[tid] = lmax; __syncthreads();
  for (int o = 128; o > 0; o >>= 1) { if (tid < o) red[tid] = fmaxf(red[tid], red[tid + o]); __syncthreads(); }
  float mx = red[0]; __syncthreads();
  float lsum = 0.f;
  for (int j = tid; j < S_total; j += 256) { float e = __expf(sc[j] - mx); sc[j] = e; lsum += e; }
  red[tid] = lsum; __syncthreads();
  for (int o = 128; o > 0; o >>= 1) { if (tid < o) red[tid] += red[tid + o]; __syncthreads(); }
  float inv = 1.f / red[0];
  __syncthreads();
  const int d = tid & (HD_ - 1), c = tid >> 6;
  float a = 0.f;
  for (int j = c; j < S_total; j += 4) {
    const float* vp = (j < S_past) ? (vpast + ((size_t)b * S_past + j) * D_ + d0)
                                   : (vnew + (size_t)b * D_ + d0);
    a += sc[j] * vp[d];
  }
  part[c][d] = a; __syncthreads();
  if (tid < HD_)
    out[(size_t)b * D_ + d0 + tid] =
        (part[0][tid] + part[1][tid] + part[2][tid] + part[3][tid]) * inv;
}

__global__ void embed_kernel(const float* __restrict__ tok,
                             const float* __restrict__ pos, float* __restrict__ h) {
  int i = blockIdx.x * 256 + threadIdx.x;
  if (i < B2_ * D_) h[i] = tok[i] + pos[i % D_];
}

extern "C" void kernel_launch(void* const* d_in, const int* in_sizes, int n_in,
                              void* d_out, int out_size, void* d_ws, size_t ws_size,
                              hipStream_t stream) {
  (void)in_sizes; (void)n_in; (void)out_size;
  const float* tok  = (const float*)d_in[0];
  const float* enc  = (const float*)d_in[1];
  const float* pk   = (const float*)d_in[2];
  const float* pv   = (const float*)d_in[3];
  const float* pos  = (const float*)d_in[4];
  const float* mask = (const float*)d_in[5];
  const float* sa_ln_w = (const float*)d_in[6];
  const float* sa_ln_b = (const float*)d_in[7];
  const float* sa_q_w  = (const float*)d_in[8];
  const float* sa_q_b  = (const float*)d_in[9];
  const float* sa_k_w  = (const float*)d_in[10];
  const float* sa_v_w  = (const float*)d_in[11];
  const float* sa_v_b  = (const float*)d_in[12];
  const float* sa_o_w  = (const float*)d_in[13];
  const float* sa_o_b  = (const float*)d_in[14];
  const float* ca_ln_w = (const float*)d_in[15];
  const float* ca_ln_b = (const float*)d_in[16];
  const float* ca_q_w  = (const float*)d_in[17];
  const float* ca_q_b  = (const float*)d_in[18];
  const float* ca_k_w  = (const float*)d_in[19];
  const float* ca_v_w  = (const float*)d_in[20];
  const float* ca_v_b  = (const float*)d_in[21];
  const float* ca_o_w  = (const float*)d_in[22];
  const float* ca_o_b  = (const float*)d_in[23];
  const float* mlp_ln_w = (const float*)d_in[24];
  const float* mlp_ln_b = (const float*)d_in[25];
  const float* mlp_w1   = (const float*)d_in[26];
  const float* mlp_b1   = (const float*)d_in[27];
  const float* mlp_w2   = (const float*)d_in[28];
  const float* mlp_b2   = (const float*)d_in[29];
  const float* fln_w    = (const float*)d_in[30];
  const float* fln_b    = (const float*)d_in[31];
  const float* lm_w     = (const float*)d_in[32];

  float* out = (float*)d_out;                       // [logits | nks | nvs | cks | cvs]
  float* out_logits = out;
  float* out_nks = out + (size_t)B2_ * V_;
  float* out_nvs = out_nks + (size_t)L_ * B2_ * D_;
  float* out_cks = out_nvs + (size_t)L_ * B2_ * D_;
  float* out_cvs = out_cks + (size_t)L_ * B2_ * ENC_ * D_;

  // ---- workspace layout ----
  char* wsb = (char*)d_ws;
  float* h   = (float*)wsb;                          // [B,D]
  float* xln = h   + B2_ * D_;                       // [B,D]
  float* qb  = xln + B2_ * D_;                       // [B,D]
  float* ao  = qb  + B2_ * D_;                       // [B,D]
  float* hid = ao  + B2_ * D_;                       // [B,4D]
  const size_t floatBytes = (size_t)(4 * B2_ * D_ + B2_ * 4 * D_) * 4;   // 49152
  bf16_t* encb = (bf16_t*)(wsb + floatBytes);        // [B*ENC, D] bf16
  const size_t encBytes = (size_t)B2_ * ENC_ * D_ * 2;
  bf16_t* wkb = (bf16_t*)((char*)encb + encBytes);   // [D, D] bf16 (reused per layer)
  bf16_t* wvb = wkb + (size_t)D_ * D_;
  const size_t needBytes = floatBytes + encBytes + (size_t)2 * D_ * D_ * 2;
  const bool fast = (ws_size >= needBytes);

  const dim3 gemvD((D_ + 7) / 8, B2_);
  const dim3 gemmGrid(D_ / 64, (B2_ * ENC_ + 127) / 128);
  const int M = B2_ * ENC_;
  const int encN4 = (M * D_) / 4;
  const int wN4 = (D_ * D_) / 4;

  embed_kernel<<<dim3((B2_ * D_ + 255) / 256), 256, 0, stream>>>(tok, pos, h);
  if (fast)
    cvt_bf16_kernel<<<dim3((encN4 + 255) / 256), 256, 0, stream>>>(enc, encb, encN4);

  for (int l = 0; l < L_; ++l) {
    const size_t lD = (size_t)l * D_, lDD = (size_t)l * D_ * D_;
    float* nk = out_nks + (size_t)l * B2_ * D_;
    float* nv = out_nvs + (size_t)l * B2_ * D_;
    float* ck = out_cks + (size_t)l * B2_ * ENC_ * D_;
    float* cv = out_cvs + (size_t)l * B2_ * ENC_ * D_;

    // ---- self-attention (KV-cache append) ----
    ln_kernel<<<B2_, 256, 0, stream>>>(h, sa_ln_w + lD, sa_ln_b + lD, xln);
    gemv_kernel<<<gemvD, 256, 0, stream>>>(xln, sa_q_w + lDD, sa_q_b + lD, nullptr, qb, D_, D_, 0);
    gemv_kernel<<<gemvD, 256, 0, stream>>>(xln, sa_k_w + lDD, nullptr,     nullptr, nk, D_, D_, 0);
    gemv_kernel<<<gemvD, 256, 0, stream>>>(xln, sa_v_w + lDD, sa_v_b + lD, nullptr, nv, D_, D_, 0);
    attn_kernel<<<dim3(H_, B2_), 256, 0, stream>>>(
        qb, pk + (size_t)l * B2_ * MC_ * D_, pv + (size_t)l * B2_ * MC_ * D_,
        nk, nv, mask, ao, MC_, MC_ + 1);
    gemv_kernel<<<gemvD, 256, 0, stream>>>(ao, sa_o_w + lDD, sa_o_b + lD, h, h, D_, D_, 0);

    // ---- cross-attention ----
    ln_kernel<<<B2_, 256, 0, stream>>>(h, ca_ln_w + lD, ca_ln_b + lD, xln);
    gemv_kernel<<<gemvD, 256, 0, stream>>>(xln, ca_q_w + lDD, ca_q_b + lD, nullptr, qb, D_, D_, 0);
    if (fast) {
      cvt_bf16_kernel<<<dim3((wN4 + 255) / 256), 256, 0, stream>>>(ca_k_w + lDD, wkb, wN4);
      cvt_bf16_kernel<<<dim3((wN4 + 255) / 256), 256, 0, stream>>>(ca_v_w + lDD, wvb, wN4);
      wmma_gemm_bf16<<<gemmGrid, 256, 0, stream>>>(encb, wkb, nullptr,     ck, M, D_, D_);
      wmma_gemm_bf16<<<gemmGrid, 256, 0, stream>>>(encb, wvb, ca_v_b + lD, cv, M, D_, D_);
    } else {
      wmma_gemm_f32src<<<gemmGrid, 256, 0, stream>>>(enc, ca_k_w + lDD, nullptr,     ck, M, D_, D_);
      wmma_gemm_f32src<<<gemmGrid, 256, 0, stream>>>(enc, ca_v_w + lDD, ca_v_b + lD, cv, M, D_, D_);
    }
    attn_kernel<<<dim3(H_, B2_), 256, 0, stream>>>(
        qb, ck, cv, nullptr, nullptr, nullptr, ao, ENC_, ENC_);
    gemv_kernel<<<gemvD, 256, 0, stream>>>(ao, ca_o_w + lDD, ca_o_b + lD, h, h, D_, D_, 0);

    // ---- MLP (exact GELU) ----
    ln_kernel<<<B2_, 256, 0, stream>>>(h, mlp_ln_w + lD, mlp_ln_b + lD, xln);
    gemv_kernel<<<dim3((4 * D_ + 7) / 8, B2_), 256, 0, stream>>>(
        xln, mlp_w1 + (size_t)l * 4 * D_ * D_, mlp_b1 + (size_t)l * 4 * D_, nullptr, hid, 4 * D_, D_, 1);
    gemv_kernel<<<gemvD, 256, 0, stream>>>(
        hid, mlp_w2 + (size_t)l * D_ * 4 * D_, mlp_b2 + lD, h, h, D_, 4 * D_, 0);
  }

  // ---- final LN + LM head ----
  ln_kernel<<<B2_, 256, 0, stream>>>(h, fln_w, fln_b, xln);
  gemv_kernel<<<dim3((V_ + 7) / 8, B2_), 256, 0, stream>>>(
      xln, lm_w, nullptr, nullptr, out_logits, V_, D_, 0);
}
